// SE3Transformer_58411555226300
// MI455X (gfx1250) — matrix operand order repaired and checked by reading, not compiled, not run on hardware
//
#include <hip/hip_runtime.h>

// ---------------- model constants ----------------
#define NDG     4
#define NCH     16
#define CKD     4
#define AFEAT   128
#define NNODES  4096
#define NEDGES  98304
#define COUT_   64
#define MAXJ    6
#define RHID    32
#define NTILES  (NEDGES/16)
#define WPB     2          // waves (16-edge tiles) per block

typedef __attribute__((ext_vector_type(16))) _Float16 v16h;
typedef __attribute__((ext_vector_type(8)))  float    v8f;
typedef _Float16 half_t;

// ================= device helpers =================
__device__ __forceinline__ double dfact(int n){ double f=1.0; for(int i=2;i<=n;++i) f*=(double)i; return f; }

__device__ double cg_coef(int j1,int m1,int j2,int m2,int j3,int m3){
  if(m1+m2!=m3) return 0.0;
  double pref = sqrt((2.0*j3+1.0)*dfact(j1+j2-j3)*dfact(j1-j2+j3)*dfact(-j1+j2+j3)/dfact(j1+j2+j3+1));
  pref *= sqrt(dfact(j3+m3)*dfact(j3-m3)*dfact(j1-m1)*dfact(j1+m1)*dfact(j2-m2)*dfact(j2+m2));
  int kmin = 0; if(j2-j3-m1>kmin) kmin=j2-j3-m1; if(j1+m2-j3>kmin) kmin=j1+m2-j3;
  int kmax = j1+j2-j3; if(j1-m1<kmax) kmax=j1-m1; if(j2+m2<kmax) kmax=j2+m2;
  double s=0.0;
  for(int k=kmin;k<=kmax;++k){
    double dnm = dfact(k)*dfact(j1+j2-j3-k)*dfact(j1-m1-k)*dfact(j2+m2-k)*dfact(j3-j2+m1+k)*dfact(j3-j1-m2+k);
    s += ((k&1)? -1.0:1.0)/dnm;
  }
  return pref*s;
}

struct cplx{ double re, im; };
__device__ __forceinline__ cplx cmul(cplx a, cplx b){ cplx r; r.re=a.re*b.re-a.im*b.im; r.im=a.re*b.im+a.im*b.re; return r; }

// nonzeros of row r of change-of-basis U(l); returns count (1 or 2)
__device__ int urow(int l,int r,int* cols, cplx* vals){
  const double s=0.7071067811865476;
  if(r==l){ cols[0]=l; vals[0]={1.0,0.0}; return 1; }
  if(r>l){ int m=r-l; double sg=(m&1)?-1.0:1.0;
    cols[0]=l+m; vals[0]={sg*s,0.0};
    cols[1]=l-m; vals[1]={s,0.0}; return 2; }
  int m=l-r;
  cols[0]=l-m; vals[0]={0.0,s};
  double sg=(m&1)?-1.0:1.0;
  cols[1]=l+m; vals[1]={0.0,-sg*s};
  return 2;
}

// ============ CG table kernel: one block per (di,do,J) ============
__global__ void se3_cg_kernel(float* __restrict__ cgbuf){
  __shared__ double qre[640], qim[640];
  __shared__ double sfac[2];
  int b = blockIdx.x;
  int DI=0, DO=0, JJ=0, base=0;
  {
    int cnt=0, pairbase=0; bool found=false;
    for(int di=0; di<NDG && !found; ++di)
      for(int dd=0; dd<NDG && !found; ++dd){
        int J0 = di>dd? di-dd : dd-di;
        int nJ = 2*(di<dd?di:dd)+1;
        int mo = 2*dd+1, mi = 2*di+1;
        for(int j=0;j<nJ;++j){
          if(cnt==b){ DI=di; DO=dd; JJ=J0+j; base = pairbase + mo*mi*(j*(2*J0+j)); found=true; break; }
          ++cnt;
        }
        if(!found) pairbase += mo*mi*(nJ*(2*J0+nJ));
      }
  }
  const int MI=2*DI+1, MO=2*DO+1, TW=2*JJ+1;
  const int T = MO*MI*TW;
  for(int t=threadIdx.x; t<T; t+=blockDim.x){
    int O = t/(MI*TW); int rem = t - O*(MI*TW); int I = rem/TW; int Jm = rem - I*TW;
    int co_[2], ci_[2], cj_[2]; cplx vo[2], vi[2], vj[2];
    int no = urow(DO, O, co_, vo);
    int ni = urow(DI, I, ci_, vi);
    int nj = urow(JJ, Jm, cj_, vj);
    cplx s = {0.0, 0.0};
    for(int a=0;a<no;++a){
      cplx va; va.re = vo[a].re; va.im = -vo[a].im;   // conj(U_do)
      for(int bb=0;bb<ni;++bb){
        cplx vab = cmul(va, vi[bb]);
        for(int c=0;c<nj;++c){
          int mo_ = co_[a]-DO, mi_ = ci_[bb]-DI, mj_ = cj_[c]-JJ;
          if(mi_ + mj_ != mo_) continue;
          double q = cg_coef(DI, mi_, JJ, mj_, DO, mo_);
          if(q==0.0) continue;
          cplx v3 = cmul(vab, vj[c]);
          s.re += v3.re*q; s.im += v3.im*q;
        }
      }
    }
    qre[t]=s.re; qim[t]=s.im;
  }
  __syncthreads();
  if(threadIdx.x==0){
    double best=-1.0; int bi=0;
    for(int t=0;t<T;++t){ double a2=qre[t]*qre[t]+qim[t]*qim[t]; if(a2>best){best=a2;bi=t;} }
    double ab = sqrt(best);
    if(ab>1e-12){ sfac[0]=qre[bi]/ab; sfac[1]=-qim[bi]/ab; }   // conj(ph)/|ph|
    else        { sfac[0]=1.0;        sfac[1]=0.0; }
  }
  __syncthreads();
  for(int t=threadIdx.x;t<T;t+=blockDim.x)
    cgbuf[base+t] = (float)(qre[t]*sfac[0] - qim[t]*sfac[1]);
}

// ============ per-edge geometry: r + spherical harmonics Ys[49] ============
__global__ void se3_geom(const float* __restrict__ pos, const int* __restrict__ srcp,
                         const int* __restrict__ dstp, float* __restrict__ rbuf,
                         float* __restrict__ ysb){
  int e=blockIdx.x*blockDim.x+threadIdx.x; if(e>=NEDGES) return;
  int s=srcp[e], d=dstp[e];
  float dx=pos[3*d]-pos[3*s], dy=pos[3*d+1]-pos[3*s+1], dz=pos[3*d+2]-pos[3*s+2];
  float r=sqrtf(dx*dx+dy*dy+dz*dz+1e-9f);
  rbuf[e]=r;
  float x=dx/r, y=dy/r, z=dz/r;
  float rho=sqrtf(fmaxf(x*x+y*y,1e-18f));
  float cphi=x/rho, sphi=y/rho;
  float cm[MAXJ+1], sm[MAXJ+1];
  cm[0]=1.f; sm[0]=0.f; cm[1]=cphi; sm[1]=sphi;
  for(int m=2;m<=MAXJ;++m){
    float c=cm[m-1]*cphi-sm[m-1]*sphi;
    float ss=sm[m-1]*cphi+cm[m-1]*sphi;
    cm[m]=c; sm[m]=ss;
  }
  float P[MAXJ+1][MAXJ+1];
  P[0][0]=1.f;
  for(int m=1;m<=MAXJ;++m) P[m][m]=P[m-1][m-1]*(-(2.f*m-1.f))*rho;
  for(int m=0;m<MAXJ;++m) P[m+1][m]=(2.f*m+1.f)*z*P[m][m];
  for(int m=0;m<=MAXJ;++m)
    for(int l=m+2;l<=MAXJ;++l)
      P[l][m]=((2.f*l-1.f)*z*P[l-1][m]-(float)(l+m-1)*P[l-2][m])/(float)(l-m);
  float* yse = ysb + (size_t)e*49;
  for(int l=0;l<=MAXJ;++l){
    for(int mm=-l;mm<=l;++mm){
      int am = mm<0? -mm: mm;
      double Kd = sqrt((2.0*l+1.0)/(4.0*3.14159265358979323846)*dfact(l-am)/dfact(l+am));
      float val;
      if(mm==0)      val = (float)Kd*P[l][0];
      else if(mm>0)  val = 1.41421356237f*(float)Kd*P[l][mm]*cm[mm];
      else           val = 1.41421356237f*(float)Kd*P[l][am]*sm[am];
      yse[l*l + (l+mm)] = val;
    }
  }
}

// ============ small utility kernels ============
__global__ void se3_cvt_half(const float* __restrict__ s, half_t* __restrict__ d, int n){
  int i=blockIdx.x*blockDim.x+threadIdx.x; if(i<n) d[i]=(half_t)s[i];
}
__global__ void se3_embed(const float* __restrict__ emb, const int* __restrict__ an, float* __restrict__ h0){
  int i=blockIdx.x*blockDim.x+threadIdx.x;
  if(i<NNODES*AFEAT){ int n=i/AFEAT, c=i-n*AFEAT; h0[i]=emb[(size_t)an[n]*AFEAT+c]; }
}
__global__ void se3_qproj(const float* __restrict__ W, const float* __restrict__ h,
                          float* __restrict__ q, int ci, int md){
  int t=blockIdx.x*blockDim.x+threadIdx.x; int tot=NNODES*CKD*md; if(t>=tot) return;
  int n=t/(CKD*md); int rem=t-n*(CKD*md); int o=rem/md; int m=rem-o*md;
  const float* hp=h+(size_t)n*ci*md;
  float s=0.f;
  for(int c=0;c<ci;++c) s+=W[o*ci+c]*hp[c*md+m];
  q[t]=s;
}
__global__ void se3_logit(const float* __restrict__ qb, const float* __restrict__ kb,
                          const int* __restrict__ dstp, float* __restrict__ lg,
                          int ndeg, float scale){
  int e=blockIdx.x*blockDim.x+threadIdx.x; if(e>=NEDGES) return;
  int n=dstp[e]; float s=0.f;
  for(int d=0;d<ndeg;++d){ int md=2*d+1;
    const float* qd = qb + (size_t)4*NNODES*d*d + (size_t)n*CKD*md;
    const float* kd = kb + (size_t)4*NEDGES*d*d + (size_t)e*CKD*md;
    for(int t=0;t<CKD*md;++t) s+=qd[t]*kd[t];
  }
  lg[e]=s*scale;
}
__global__ void se3_attmax(const float* __restrict__ lg, const int* __restrict__ dstp,
                           unsigned* __restrict__ amax){
  int e=blockIdx.x*blockDim.x+threadIdx.x; if(e>=NEDGES) return;
  unsigned u=__float_as_uint(lg[e]);
  unsigned enc=(u&0x80000000u)? ~u : (u|0x80000000u);   // monotonic float->uint
  atomicMax(&amax[dstp[e]],enc);
}
__global__ void se3_attsum(const float* __restrict__ lg, const int* __restrict__ dstp,
                           const unsigned* __restrict__ amax, float* __restrict__ ev,
                           float* __restrict__ asum){
  int e=blockIdx.x*blockDim.x+threadIdx.x; if(e>=NEDGES) return;
  unsigned u=amax[dstp[e]];
  float m=(u&0x80000000u)? __uint_as_float(u&0x7fffffffu) : __uint_as_float(~u);
  float x=expf(lg[e]-m); ev[e]=x; atomicAdd(&asum[dstp[e]],x);
}
__global__ void se3_agg(const float* __restrict__ ev, const float* __restrict__ asum,
                        const float* __restrict__ vb, const int* __restrict__ dstp,
                        float* __restrict__ agg){
  int e=blockIdx.x*blockDim.x+threadIdx.x; if(e>=NEDGES) return;
  int n=dstp[e];
  float a = ev[e]/(asum[n]+1e-9f);
  for(int d=0;d<NDG;++d){ int md=2*d+1;
    const float* vd = vb + (size_t)4*NEDGES*d*d + (size_t)e*CKD*md;
    float* ad = agg + (size_t)4*NNODES*d*d + (size_t)n*CKD*md;
    for(int t=0;t<CKD*md;++t) atomicAdd(&ad[t], a*vd[t]);
  }
}
__global__ void se3_proj(const float* __restrict__ P, const float* __restrict__ agg,
                         const float* __restrict__ hold, float* __restrict__ hnew,
                         int md, int ciold){
  int cin=CKD+ciold;
  int t=blockIdx.x*blockDim.x+threadIdx.x; int tot=NNODES*NCH*md; if(t>=tot) return;
  int n=t/(NCH*md); int rem=t-n*(NCH*md); int o=rem/md; int m=rem-o*md;
  const float* ap = agg + (size_t)n*CKD*md;
  float s=0.f;
  for(int c=0;c<CKD;++c) s+=P[o*cin+c]*ap[c*md+m];
  if(ciold>0){
    const float* hp = hold + (size_t)n*ciold*md;
    for(int c=0;c<ciold;++c) s+=P[o*cin+CKD+c]*hp[c*md+m];
  }
  hnew[t]=s;
}
__global__ void se3_gnorm(float* __restrict__ h, const float* __restrict__ gamma,
                          const float* __restrict__ beta, int md){
  int n=blockIdx.x*blockDim.x+threadIdx.x; if(n>=NNODES) return;
  float* hp = h + (size_t)n*NCH*md;
  float nrm[NCH]; float mu=0.f;
  for(int c=0;c<NCH;++c){
    float s=0.f;
    for(int m=0;m<md;++m){ float v=hp[c*md+m]; s+=v*v; }
    nrm[c]=sqrtf(s+1e-12f); mu+=nrm[c];
  }
  mu *= (1.f/NCH);
  float var=0.f;
  for(int c=0;c<NCH;++c){ float d=nrm[c]-mu; var+=d*d; }
  var *= (1.f/NCH);
  float inv = rsqrtf(var+1e-5f);
  for(int c=0;c<NCH;++c){
    float sc = fmaxf(gamma[c]*(nrm[c]-mu)*inv+beta[c],0.f)/nrm[c];
    for(int m=0;m<md;++m) hp[c*md+m]*=sc;
  }
}
__global__ void se3_coagg(const float* __restrict__ msg, const int* __restrict__ dstp,
                          float* __restrict__ out0){
  int e=blockIdx.x*blockDim.x+threadIdx.x; if(e>=NEDGES) return;
  int n=dstp[e];
  const float* mp = msg + (size_t)e*COUT_;
  for(int o=0;o<COUT_;++o) atomicAdd(&out0[(size_t)n*COUT_+o], mp[o]);
}
__global__ void se3_coself(const float* __restrict__ sw, const float* __restrict__ h0,
                           float* __restrict__ out0){
  int t=blockIdx.x*blockDim.x+threadIdx.x; if(t>=NNODES*COUT_) return;
  int n=t/COUT_, o=t-n*COUT_;
  float s=0.f;
  for(int c=0;c<NCH;++c) s+=sw[o*NCH+c]*h0[(size_t)n*NCH+c];
  out0[t]+=s;
}
__global__ void se3_pool(const float* __restrict__ out0, float* __restrict__ pooled){
  int t=blockIdx.x*blockDim.x+threadIdx.x; if(t>=NNODES*COUT_) return;
  atomicAdd(&pooled[t%COUT_], out0[t]*(1.0f/NNODES));
}
__global__ void se3_head(const float* __restrict__ pooled, const float* __restrict__ w1,
                         const float* __restrict__ b1, const float* __restrict__ w2,
                         const float* __restrict__ b2, float* __restrict__ out){
  __shared__ float z[COUT_];
  int o=threadIdx.x;
  if(o<COUT_){
    float s=b1[o];
    for(int c=0;c<COUT_;++c) s+=w1[o*COUT_+c]*pooled[c];
    z[o]=fmaxf(s,0.f);
  }
  __syncthreads();
  if(o==0){
    float s=b2[0];
    for(int c=0;c<COUT_;++c) s+=w2[c]*z[c];
    out[0]=s;
  }
}

// ============ fused radial-MLP (WMMA) + tensor-field conv kernel ============
// Per wave: 16 edges. Radial GEMM (16 edges x 32 hid) x (32 x 16 R-entries) via
// V_WMMA_F32_16X16X32_F16, result staged through LDS, then CG-basis contraction.
// All dynamically-indexed scratch arrays live in LDS -> kernel is scratch-free
// (avoids backend FI-elimination register scavenging failure).
template<int CI,int DI,int DO,int CO>
__global__ __launch_bounds__(64)
void se3_econv(const float* __restrict__ h,   const int* __restrict__ srcp,
               const float* __restrict__ rr,  const float* __restrict__ ysb,
               const float* __restrict__ w1,  const float* __restrict__ b1,
               const half_t* __restrict__ w2h,const float* __restrict__ b2,
               const float* __restrict__ cg,  float* __restrict__ outp, int accumulate)
{
  constexpr int MI  = 2*DI+1;
  constexpr int MO  = 2*DO+1;
  constexpr int MN  = (DI<DO?DI:DO);
  constexpr int NJ  = 2*MN+1;
  constexpr int J0  = (DI>DO?DI-DO:DO-DI);
  constexpr int CPO = CI/16;        // chunks per output channel
  constexpr int CPJ = CO*CPO;       // chunks per J-slice
  constexpr int COH = CO/2;         // output channels per half-lane

  __shared__ float sh_h  [WPB*16*CI*MI];
  __shared__ float sh_ys [WPB*16*49];
  __shared__ float sh_R  [WPB*16*16];
  __shared__ float sh_bas[WPB*16*MO*MI];
  __shared__ float sh_acc[WPB*32*COH*MO];
  __shared__ int   sh_src[WPB*16];

  const int tid=threadIdx.x, wv=tid>>5, ln=tid&31;
  const int le=ln&15, hi=ln>>4;
  const int e0=(blockIdx.x*WPB+wv)*16;

  __builtin_prefetch(w2h, 0, 1);

  if(ln<16) sh_src[wv*16+ln]=srcp[e0+ln];
  __syncthreads();
  for(int t=ln;t<16*CI*MI;t+=32){
    int ee=t/(CI*MI), off=t-ee*(CI*MI);
    sh_h[(wv*16+ee)*(CI*MI)+off]=h[(size_t)sh_src[wv*16+ee]*(CI*MI)+off];
  }
  for(int t=ln;t<16*49;t+=32){
    int ee=t/49, off=t-ee*49;
    sh_ys[(wv*16+ee)*49+off]=ysb[(size_t)(e0+ee)*49+off];
  }
  float* accp = &sh_acc[(wv*32+ln)*COH*MO];   // per-lane accumulator slot
  for(int t=0;t<COH*MO;++t) accp[t]=0.f;
  __syncthreads();

  // A-fragment: hid = relu(w1*r+b1), packed in contiguous K order (K = 16*hi + kk)
  float rv=rr[e0+le];
  v16h afrag;
  #pragma unroll
  for(int kk=0;kk<16;++kk){
    int K = kk + 16*hi;              // lanes 0-15: K 0..15 ; lanes 16-31: 16..31
    float hv=fmaxf(w1[K]*rv+b1[K],0.f);
    afrag[kk]=(_Float16)hv;
  }

  const float* basp = &sh_bas[(wv*16+le)*MO*MI];
  const float* hbase= &sh_h [(wv*16+le)*(CI*MI)];

  for(int j=0;j<NJ;++j){
    const int TWJ = 2*(J0+j)+1;
    const float* cgj = cg + MO*MI*(j*(2*J0+j));
    const int ybase = (J0+j)*(J0+j);
    // cooperative basis build: bas[edge][n,m] = sum_k CG[n,m,k] * Ys[J0+j][k]
    for(int t=ln;t<16*MO*MI;t+=32){
      int ee=t/(MO*MI), nm=t-ee*(MO*MI);
      const float* yj=&sh_ys[(wv*16+ee)*49+ybase];
      float s=0.f;
      for(int k=0;k<TWJ;++k) s+=cgj[nm*TWJ+k]*yj[k];
      sh_bas[(wv*16+ee)*MO*MI+nm]=s;
    }
    __syncthreads();
    #pragma unroll 1
    for(int cc=0;cc<CPJ;++cc){
      const int o  = cc/CPO;
      const int i0 = (cc-o*CPO)*16;
      const int cb = (j*CPJ+cc)*16;
      // B-fragment: lane = column (R-entry), K-half by lane group
      v16h bfrag = *(const v16h*)(w2h + (size_t)(cb+le)*RHID + hi*16);
      v8f cacc;
      float bias = b2[cb+le];
      #pragma unroll
      for(int g=0;g<8;++g) cacc[g]=bias;
      cacc = __builtin_amdgcn_wmma_f32_16x16x32_f16(false, afrag, false, bfrag,
                                                    (short)0, cacc, false, false);
      // C layout: lane holds N=(ln&15)=entry, rows M=g+8*hi = edge
      #pragma unroll
      for(int g=0;g<8;++g) sh_R[(wv*16+le)*16 + g + 8*hi] = cacc[g];
      __syncthreads();
      if(o/COH==hi){
        const int ol=o-hi*COH;
        float* ap = accp + ol*MO;
        #pragma unroll 1
        for(int i=0;i<16;++i){
          float Rv = sh_R[(wv*16+i)*16+le];
          const float* hrow = hbase + (i0+i)*MI;
          for(int n=0;n<MO;++n){
            float ts=0.f;
            for(int m=0;m<MI;++m) ts += basp[n*MI+m]*hrow[m];
            ap[n] += Rv*ts;
          }
        }
      }
      __syncthreads();
    }
  }
  float* op = outp + ((size_t)(e0+le)*CO + (size_t)hi*COH)*MO;
  if(accumulate){ for(int t=0;t<COH*MO;++t) op[t]+=accp[t]; }
  else          { for(int t=0;t<COH*MO;++t) op[t] =accp[t]; }
}

// ================= host side =================
struct RadialP { const float *b1,*b2,*w1,*w2; };

extern "C" void kernel_launch(void* const* d_in, const int* in_sizes, int n_in,
                              void* d_out, int out_size, void* d_ws, size_t ws_size,
                              hipStream_t stream)
{
  (void)in_sizes; (void)n_in; (void)out_size; (void)ws_size;
  // ---- walk params: JAX pytree flatten (sorted dict keys), top-level insertion order ----
  int ix=0;
  auto F=[&]()->const float*{ return (const float*)d_in[ix++]; };

  RadialP co_rad[4];                                     // conv_out.radial '0,0'..'3,0'
  for(int di=0;di<4;++di){ co_rad[di].b1=F(); co_rad[di].b2=F(); co_rad[di].w1=F(); co_rad[di].w2=F(); }
  const float* self_w=F();                               // conv_out.self_w
  const float* embed =F();                               // embed
  const float* fc1_b=F(); const float* fc1_w=F();
  const float* fc2_b=F(); const float* fc2_w=F();
  // layer 0: k, norm, proj, q, v
  RadialP l0_k; l0_k.b1=F(); l0_k.b2=F(); l0_k.w1=F(); l0_k.w2=F();
  const float *l0_beta[4], *l0_gamma[4];
  for(int d=0;d<4;++d){ l0_beta[d]=F(); l0_gamma[d]=F(); }
  const float* l0_proj[4]; for(int d=0;d<4;++d) l0_proj[d]=F();
  const float* l0_q=F();
  RadialP l0_v[4];
  for(int o=0;o<4;++o){ l0_v[o].b1=F(); l0_v[o].b2=F(); l0_v[o].w1=F(); l0_v[o].w2=F(); }
  // layer 1: k (di-major sorted), norm, proj, q, v
  RadialP l1_k[4][4];
  for(int di=0;di<4;++di) for(int o=0;o<4;++o){ l1_k[di][o].b1=F(); l1_k[di][o].b2=F(); l1_k[di][o].w1=F(); l1_k[di][o].w2=F(); }
  const float *l1_beta[4], *l1_gamma[4];
  for(int d=0;d<4;++d){ l1_beta[d]=F(); l1_gamma[d]=F(); }
  const float* l1_proj[4]; for(int d=0;d<4;++d) l1_proj[d]=F();
  const float* l1_q[4];    for(int d=0;d<4;++d) l1_q[d]=F();
  RadialP l1_v[4][4];
  for(int di=0;di<4;++di) for(int o=0;o<4;++o){ l1_v[di][o].b1=F(); l1_v[di][o].b2=F(); l1_v[di][o].w1=F(); l1_v[di][o].w2=F(); }
  const float* pos = F();
  const int* an = (const int*)d_in[ix++];
  const int* ei = (const int*)d_in[ix++];
  const int* d_src = ei;
  const int* d_dst = ei + NEDGES;

  // ---- CG table offsets (matches device se3_cg_kernel ordering) ----
  int cg_off[4][4]; int cg_total=0;
  for(int di=0;di<4;++di) for(int o=0;o<4;++o){
    cg_off[di][o]=cg_total;
    int J0 = di>o?di-o:o-di; int nJ=2*(di<o?di:o)+1;
    cg_total += (2*o+1)*(2*di+1)*(nJ*(2*J0+nJ));
  }

  // ---- workspace carve-out (~90 MB) ----
  char* wsb=(char*)d_ws; size_t wo=0;
  auto walloc=[&](size_t bytes)->void*{ wo=(wo+255)&~(size_t)255; void* p=wsb+wo; wo+=bytes; return p; };
  float*    cgbuf =(float*)walloc(sizeof(float)*cg_total);
  float*    ysbuf =(float*)walloc(sizeof(float)*(size_t)NEDGES*49);
  float*    rbuf  =(float*)walloc(sizeof(float)*NEDGES);
  float*    h0buf =(float*)walloc(sizeof(float)*(size_t)NNODES*AFEAT);
  float*    hA    =(float*)walloc(sizeof(float)*(size_t)NNODES*256);
  float*    hB    =(float*)walloc(sizeof(float)*(size_t)NNODES*256);
  float*    qbuf  =(float*)walloc(sizeof(float)*(size_t)NNODES*64);
  float*    kbuf  =(float*)walloc(sizeof(float)*(size_t)NEDGES*64);   // reused as conv_out msgs
  float*    vbuf  =(float*)walloc(sizeof(float)*(size_t)NEDGES*64);
  float*    lgbuf =(float*)walloc(sizeof(float)*NEDGES);
  float*    evbuf =(float*)walloc(sizeof(float)*NEDGES);
  unsigned* amax  =(unsigned*)walloc(sizeof(unsigned)*NNODES);
  float*    asum  =(float*)walloc(sizeof(float)*NNODES);
  float*    aggbuf=(float*)walloc(sizeof(float)*(size_t)NNODES*64);
  float*    out0  =(float*)walloc(sizeof(float)*(size_t)NNODES*COUT_);
  float*    pooled=(float*)walloc(sizeof(float)*COUT_);
  auto halloc=[&](int nout)->half_t*{ return (half_t*)walloc(sizeof(half_t)*(size_t)nout*RHID); };
  half_t* w2h_l0k = halloc(512);
  half_t* w2h_l0v[4]; for(int o=0;o<4;++o) w2h_l0v[o]=halloc(512);
  half_t* w2h_l1k[4][4]; half_t* w2h_l1v[4][4];
  for(int di=0;di<4;++di) for(int o=0;o<4;++o){
    int nJ=2*(di<o?di:o)+1;
    w2h_l1k[di][o]=halloc(nJ*64);
    w2h_l1v[di][o]=halloc(nJ*64);
  }
  half_t* w2h_co[4]; for(int di=0;di<4;++di) w2h_co[di]=halloc(1024);

  float* hAd[4]; float* hBd[4];
  for(int d=0;d<4;++d){ hAd[d]=hA+(size_t)16*NNODES*d*d; hBd[d]=hB+(size_t)16*NNODES*d*d; }
  auto kvo=[&](int d)->size_t{ return (size_t)4*NEDGES*d*d; };
  auto noo=[&](int d)->size_t{ return (size_t)4*NNODES*d*d; };

  const int EG = NTILES/WPB;

  // ---- precompute: CG tables + f16 weight images ----
  se3_cg_kernel<<<44,128,0,stream>>>(cgbuf);
  auto cvt=[&](const float* s, half_t* dsth, int n){
    se3_cvt_half<<<(n+255)/256,256,0,stream>>>(s,dsth,n);
  };
  cvt(l0_k.w2, w2h_l0k, 512*RHID);
  for(int o=0;o<4;++o) cvt(l0_v[o].w2, w2h_l0v[o], 512*RHID);
  for(int di=0;di<4;++di) for(int o=0;o<4;++o){
    int nJ=2*(di<o?di:o)+1;
    cvt(l1_k[di][o].w2, w2h_l1k[di][o], nJ*64*RHID);
    cvt(l1_v[di][o].w2, w2h_l1v[di][o], nJ*64*RHID);
  }
  for(int di=0;di<4;++di) cvt(co_rad[di].w2, w2h_co[di], 1024*RHID);

  se3_embed<<<(NNODES*AFEAT+255)/256,256,0,stream>>>(embed,an,h0buf);
  se3_geom<<<(NEDGES+255)/256,256,0,stream>>>(pos,d_src,d_dst,rbuf,ysbuf);

  // ---- layer-2/conv econv dispatchers (compile-time (di,do)) ----
  auto econv_l2=[&](int di,int o,const RadialP& R,const half_t* w2h,float* op,int acc2){
    switch(di*4+o){
#define C2(DI_,DO_) case DI_*4+DO_: se3_econv<16,DI_,DO_,4><<<EG,64,0,stream>>>(hAd[DI_],d_src,rbuf,ysbuf,R.w1,R.b1,w2h,R.b2,cgbuf+cg_off[DI_][DO_],op,acc2); break;
      C2(0,0) C2(0,1) C2(0,2) C2(0,3)
      C2(1,0) C2(1,1) C2(1,2) C2(1,3)
      C2(2,0) C2(2,1) C2(2,2) C2(2,3)
      C2(3,0) C2(3,1) C2(3,2) C2(3,3)
#undef C2
    }
  };
  auto econv_co=[&](int di,const RadialP& R,const half_t* w2h,float* op,int acc2){
    switch(di){
#define C3(DI_) case DI_: se3_econv<16,DI_,0,64><<<EG,64,0,stream>>>(hBd[DI_],d_src,rbuf,ysbuf,R.w1,R.b1,w2h,R.b2,cgbuf+cg_off[DI_][0],op,acc2); break;
      C3(0) C3(1) C3(2) C3(3)
#undef C3
    }
  };

  // ================= layer 1 (in_fiber = {0:128}) =================
  se3_qproj<<<(NNODES*4+255)/256,256,0,stream>>>(l0_q,h0buf,qbuf,AFEAT,1);
  se3_econv<128,0,0,4><<<EG,64,0,stream>>>(h0buf,d_src,rbuf,ysbuf,l0_k.w1,l0_k.b1,w2h_l0k,l0_k.b2,cgbuf+cg_off[0][0],kbuf,0);
  se3_econv<128,0,0,4><<<EG,64,0,stream>>>(h0buf,d_src,rbuf,ysbuf,l0_v[0].w1,l0_v[0].b1,w2h_l0v[0],l0_v[0].b2,cgbuf+cg_off[0][0],vbuf+kvo(0),0);
  se3_econv<128,0,1,4><<<EG,64,0,stream>>>(h0buf,d_src,rbuf,ysbuf,l0_v[1].w1,l0_v[1].b1,w2h_l0v[1],l0_v[1].b2,cgbuf+cg_off[0][1],vbuf+kvo(1),0);
  se3_econv<128,0,2,4><<<EG,64,0,stream>>>(h0buf,d_src,rbuf,ysbuf,l0_v[2].w1,l0_v[2].b1,w2h_l0v[2],l0_v[2].b2,cgbuf+cg_off[0][2],vbuf+kvo(2),0);
  se3_econv<128,0,3,4><<<EG,64,0,stream>>>(h0buf,d_src,rbuf,ysbuf,l0_v[3].w1,l0_v[3].b1,w2h_l0v[3],l0_v[3].b2,cgbuf+cg_off[0][3],vbuf+kvo(3),0);
  se3_logit<<<(NEDGES+255)/256,256,0,stream>>>(qbuf,kbuf,d_dst,lgbuf,1,0.5f);   // 1/sqrt(4)
  (void)hipMemsetAsync(amax,0,sizeof(unsigned)*NNODES,stream);
  (void)hipMemsetAsync(asum,0,sizeof(float)*NNODES,stream);
  (void)hipMemsetAsync(aggbuf,0,sizeof(float)*(size_t)NNODES*64,stream);
  se3_attmax<<<(NEDGES+255)/256,256,0,stream>>>(lgbuf,d_dst,amax);
  se3_attsum<<<(NEDGES+255)/256,256,0,stream>>>(lgbuf,d_dst,amax,evbuf,asum);
  se3_agg<<<(NEDGES+255)/256,256,0,stream>>>(evbuf,asum,vbuf,d_dst,aggbuf);
  se3_proj<<<(NNODES*NCH+255)/256,256,0,stream>>>(l0_proj[0],aggbuf+noo(0),h0buf,hAd[0],1,AFEAT);
  for(int d=1;d<4;++d){ int md=2*d+1;
    se3_proj<<<(NNODES*NCH*md+255)/256,256,0,stream>>>(l0_proj[d],aggbuf+noo(d),nullptr,hAd[d],md,0); }
  for(int d=0;d<4;++d)
    se3_gnorm<<<(NNODES+255)/256,256,0,stream>>>(hAd[d],l0_gamma[d],l0_beta[d],2*d+1);

  // ================= layer 2 (in_fiber = {0..3:16}) =================
  for(int d=0;d<4;++d){ int md=2*d+1;
    se3_qproj<<<(NNODES*4*md+255)/256,256,0,stream>>>(l1_q[d],hAd[d],qbuf+noo(d),NCH,md); }
  for(int o=0;o<4;++o) for(int di=0;di<4;++di){
    econv_l2(di,o,l1_k[di][o],w2h_l1k[di][o],kbuf+kvo(o),di>0?1:0);
    econv_l2(di,o,l1_v[di][o],w2h_l1v[di][o],vbuf+kvo(o),di>0?1:0);
  }
  se3_logit<<<(NEDGES+255)/256,256,0,stream>>>(qbuf,kbuf,d_dst,lgbuf,4,0.125f);  // 1/sqrt(64)
  (void)hipMemsetAsync(amax,0,sizeof(unsigned)*NNODES,stream);
  (void)hipMemsetAsync(asum,0,sizeof(float)*NNODES,stream);
  (void)hipMemsetAsync(aggbuf,0,sizeof(float)*(size_t)NNODES*64,stream);
  se3_attmax<<<(NEDGES+255)/256,256,0,stream>>>(lgbuf,d_dst,amax);
  se3_attsum<<<(NEDGES+255)/256,256,0,stream>>>(lgbuf,d_dst,amax,evbuf,asum);
  se3_agg<<<(NEDGES+255)/256,256,0,stream>>>(evbuf,asum,vbuf,d_dst,aggbuf);
  for(int d=0;d<4;++d){ int md=2*d+1;
    se3_proj<<<(NNODES*NCH*md+255)/256,256,0,stream>>>(l1_proj[d],aggbuf+noo(d),hAd[d],hBd[d],md,NCH); }
  for(int d=0;d<4;++d)
    se3_gnorm<<<(NNODES+255)/256,256,0,stream>>>(hBd[d],l1_gamma[d],l1_beta[d],2*d+1);

  // ================= output conv + pooling + head =================
  for(int di=0;di<4;++di) econv_co(di,co_rad[di],w2h_co[di],kbuf,di>0?1:0);      // kbuf reused as msgs
  (void)hipMemsetAsync(out0,0,sizeof(float)*(size_t)NNODES*COUT_,stream);
  se3_coagg<<<(NEDGES+255)/256,256,0,stream>>>(kbuf,d_dst,out0);
  se3_coself<<<(NNODES*COUT_+255)/256,256,0,stream>>>(self_w,hBd[0],out0);
  (void)hipMemsetAsync(pooled,0,sizeof(float)*COUT_,stream);
  se3_pool<<<(NNODES*COUT_+255)/256,256,0,stream>>>(out0,pooled);
  se3_head<<<1,64,0,stream>>>(pooled,fc1_w,fc1_b,fc2_w,fc2_b,(float*)d_out);
}
